// MoEFFN_8564164788846
// MI455X (gfx1250) — matrix operand order repaired and checked by reading, not compiled, not run on hardware
//
#include <hip/hip_runtime.h>
#include <cstdint>

#define N_TOK 4096
#define DM    1024
#define DE    2048
#define NEXP  8

typedef __bf16 bf16_t;
typedef __attribute__((ext_vector_type(16))) __bf16 v16bf;
typedef __attribute__((ext_vector_type(4)))  __bf16 bf16x4;
typedef __attribute__((ext_vector_type(8)))  float  v8f;

#define BM   128
#define BN   64
#define BK   32
#define LDSA 48   // row stride in bf16 (96B, multiple of 16B)
#define LDSB 48

// ---------------- 1) router: one wave32 per token ----------------
__global__ __launch_bounds__(256) void moe_router(
    const float* __restrict__ x, const float* __restrict__ gw,
    const float* __restrict__ gb, int* __restrict__ sel, float* __restrict__ selw)
{
  const int lane = threadIdx.x & 31;
  const int t = blockIdx.x * 8 + (threadIdx.x >> 5);
  if (t >= N_TOK) return;
  float acc[NEXP];
#pragma unroll
  for (int e = 0; e < NEXP; ++e) acc[e] = 0.f;
  const float* xr = x + (size_t)t * DM;
  for (int d = lane; d < DM; d += 32) {
    float xv = xr[d];
    const float4* g4 = reinterpret_cast<const float4*>(gw + (size_t)d * NEXP);
    float4 a = g4[0], b = g4[1];
    acc[0] += xv * a.x; acc[1] += xv * a.y; acc[2] += xv * a.z; acc[3] += xv * a.w;
    acc[4] += xv * b.x; acc[5] += xv * b.y; acc[6] += xv * b.z; acc[7] += xv * b.w;
  }
#pragma unroll
  for (int s = 16; s > 0; s >>= 1)
#pragma unroll
    for (int e = 0; e < NEXP; ++e) acc[e] += __shfl_xor(acc[e], s, 32);
  if (lane == 0) {
    float l[NEXP], m = -1e30f;
#pragma unroll
    for (int e = 0; e < NEXP; ++e) { l[e] = acc[e] + gb[e]; m = fmaxf(m, l[e]); }
    float s = 0.f, p[NEXP];
#pragma unroll
    for (int e = 0; e < NEXP; ++e) { p[e] = __expf(l[e] - m); s += p[e]; }
    float inv = 1.f / s;
#pragma unroll
    for (int e = 0; e < NEXP; ++e) p[e] *= inv;
    int i1 = 0; float p1 = p[0];
#pragma unroll
    for (int e = 1; e < NEXP; ++e) if (p[e] > p1) { p1 = p[e]; i1 = e; }
    int i2 = -1; float p2 = -1.f;
#pragma unroll
    for (int e = 0; e < NEXP; ++e) if (e != i1 && p[e] > p2) { p2 = p[e]; i2 = e; }
    float nrm = 1.f / (p1 + p2 + 1e-8f);
    sel[2 * t + 0] = i1; selw[2 * t + 0] = p1 * nrm;
    sel[2 * t + 1] = i2; selw[2 * t + 1] = p2 * nrm;
  }
}

// ---------------- 2) per-expert counts + exclusive offsets ----------------
__global__ __launch_bounds__(256) void moe_count(
    const int* __restrict__ sel, int* __restrict__ counts, int* __restrict__ offsets)
{
  __shared__ int c[NEXP];
  const int tid = threadIdx.x;
  if (tid < NEXP) c[tid] = 0;
  __syncthreads();
  for (int t = tid; t < N_TOK; t += 256) {
    atomicAdd(&c[sel[2 * t]], 1);
    atomicAdd(&c[sel[2 * t + 1]], 1);
  }
  __syncthreads();
  if (tid == 0) {
    int off = 0;
#pragma unroll
    for (int e = 0; e < NEXP; ++e) { counts[e] = c[e]; offsets[e] = off; off += c[e]; }
  }
}

// ---------------- 3) deterministic token gather (1 WG per expert) ----------------
__global__ __launch_bounds__(256) void moe_gather(
    const int* __restrict__ sel, const float* __restrict__ selw,
    const int* __restrict__ offsets,
    int* __restrict__ tokList, float* __restrict__ wArena, int* __restrict__ rowOf)
{
  const int e = blockIdx.x;
  const int tid = threadIdx.x;
  __shared__ int scan[256];
  __shared__ int sBase;
  if (tid == 0) sBase = offsets[e];
  __syncthreads();
  for (int c0 = 0; c0 < N_TOK; c0 += 256) {
    const int t = c0 + tid;
    const int s0 = sel[2 * t], s1 = sel[2 * t + 1];
    int flag = 0, k = 0;
    if (s0 == e) { flag = 1; k = 0; }
    else if (s1 == e) { flag = 1; k = 1; }
    scan[tid] = flag;
    __syncthreads();
    for (int off = 1; off < 256; off <<= 1) {
      int v = (tid >= off) ? scan[tid - off] : 0;
      __syncthreads();
      scan[tid] += v;
      __syncthreads();
    }
    const int pos = sBase + scan[tid] - flag;
    if (flag) {
      tokList[pos] = t;
      wArena[pos] = selw[2 * t + k];
      rowOf[2 * t + k] = pos;
    }
    __syncthreads();
    if (tid == 0) sBase += scan[255];
    __syncthreads();
  }
}

// ---------------- WMMA fragment helpers ----------------
__device__ __forceinline__ v16bf frag_a(const bf16_t* As, int row, int lane) {
  // A 16x32 bf16: lanes<16 hold K {0..7,16..23}; lanes>=16 hold {8..15,24..31}
  const int kh = (lane >> 4) * 8;
  const bf16_t* p = As + row * LDSA + kh;
  v16bf a;
  int4* d = reinterpret_cast<int4*>(&a);
  d[0] = *reinterpret_cast<const int4*>(p);
  d[1] = *reinterpret_cast<const int4*>(p + 16);
  return a;
}
__device__ __forceinline__ v16bf frag_b(const bf16_t* Bs, int nrow, int lane) {
  // B 32x16 bf16: lanes<16 hold K {0..7,16..23}; lanes>=16 hold {8..15,24..31}
  const int kb = (lane >> 4) * 8;
  const bf16_t* p = Bs + nrow * LDSB + kb;
  v16bf b;
  int4* d = reinterpret_cast<int4*>(&b);
  d[0] = *reinterpret_cast<const int4*>(p);
  d[1] = *reinterpret_cast<const int4*>(p + 16);
  return b;
}

// ---------------- 4) grouped GEMM1: hidden = silu(x@Wg) * (x@Wv), bf16 out ----------------
__global__ __launch_bounds__(256) void moe_gemm1(
    const float* __restrict__ x, const float* __restrict__ wgate,
    const float* __restrict__ wval, const int* __restrict__ tokList,
    const int* __restrict__ counts, const int* __restrict__ offsets,
    bf16_t* __restrict__ hidden)
{
  const int e = blockIdx.z;
  const int cnt = counts[e];
  const int m0 = blockIdx.x * BM;
  if (m0 >= cnt) return;
  const int n0 = blockIdx.y * BN;
  const int base = offsets[e];
  const float* wg = wgate + (size_t)e * DM * DE;
  const float* wv = wval + (size_t)e * DM * DE;

  __shared__ bf16_t As[BM * LDSA];
  __shared__ bf16_t Bg[BN * LDSB];
  __shared__ bf16_t Bv[BN * LDSB];
  __shared__ int sTok[BM];

  const int tid = threadIdx.x;
  if (tid < BM) {
    int m = m0 + tid;
    sTok[tid] = tokList[base + (m < cnt ? m : cnt - 1)];
  }
  __syncthreads();

  const int lane = tid & 31;
  const int wid = tid >> 5;
  const int wm = (wid & 3) * 32;   // 4 waves along M
  const int wn = (wid >> 2) * 32;  // 2 waves along N
  const int lrow = lane & 15;

  v8f cg[2][2], cv[2][2];
#pragma unroll
  for (int i = 0; i < 2; ++i)
#pragma unroll
    for (int j = 0; j < 2; ++j)
#pragma unroll
      for (int q = 0; q < 8; ++q) { cg[i][j][q] = 0.f; cv[i][j][q] = 0.f; }

  for (int k0 = 0; k0 < DM; k0 += BK) {
    { // stage A: gathered fp32 token rows -> bf16 LDS
      const int row = tid >> 1;
      const int kq = (tid & 1) * 16;
      const float4* src = reinterpret_cast<const float4*>(
          x + (size_t)sTok[row] * DM + k0 + kq);
#pragma unroll
      for (int q = 0; q < 4; ++q) {
        float4 f = src[q];
        bf16x4 h;
        h[0] = (bf16_t)f.x; h[1] = (bf16_t)f.y; h[2] = (bf16_t)f.z; h[3] = (bf16_t)f.w;
        *reinterpret_cast<bf16x4*>(&As[row * LDSA + kq + q * 4]) = h;
      }
    }
#pragma unroll
    for (int i = 0; i < 2; ++i) { // stage Bg/Bv: fp32 [K,N] -> bf16 [N][K] LDS
      const int u = tid + 256 * i;
      const int n = u & 63;
      const int k = (u >> 6) * 4;
      const size_t gidx = (size_t)(k0 + k) * DE + (n0 + n);
      bf16x4 hg, hv;
#pragma unroll
      for (int q = 0; q < 4; ++q) {
        hg[q] = (bf16_t)wg[gidx + (size_t)q * DE];
        hv[q] = (bf16_t)wv[gidx + (size_t)q * DE];
      }
      *reinterpret_cast<bf16x4*>(&Bg[n * LDSB + k]) = hg;
      *reinterpret_cast<bf16x4*>(&Bv[n * LDSB + k]) = hv;
    }
    __syncthreads();

    v16bf a[2], bg[2], bv[2];
#pragma unroll
    for (int mi = 0; mi < 2; ++mi) a[mi] = frag_a(As, wm + mi * 16 + lrow, lane);
#pragma unroll
    for (int ni = 0; ni < 2; ++ni) {
      bg[ni] = frag_b(Bg, wn + ni * 16 + lrow, lane);
      bv[ni] = frag_b(Bv, wn + ni * 16 + lrow, lane);
    }
#pragma unroll
    for (int mi = 0; mi < 2; ++mi)
#pragma unroll
      for (int ni = 0; ni < 2; ++ni) {
        cg[mi][ni] = __builtin_amdgcn_wmma_f32_16x16x32_bf16(
            false, a[mi], false, bg[ni], (short)0, cg[mi][ni], false, false);
        cv[mi][ni] = __builtin_amdgcn_wmma_f32_16x16x32_bf16(
            false, a[mi], false, bv[ni], (short)0, cv[mi][ni], false, false);
      }
    __syncthreads();
  }

  const int mhi = (lane >> 4) * 8;
#pragma unroll
  for (int mi = 0; mi < 2; ++mi)
#pragma unroll
    for (int ni = 0; ni < 2; ++ni) {
      const int col = n0 + wn + ni * 16 + lrow;
#pragma unroll
      for (int j = 0; j < 8; ++j) {
        const int m = wm + mi * 16 + mhi + j;
        if (m0 + m < cnt) {
          float g = cg[mi][ni][j];
          float v = cv[mi][ni][j];
          float h = (g / (1.f + __expf(-g))) * v; // silu(g) * v
          hidden[(size_t)(base + m0 + m) * DE + col] = (bf16_t)h;
        }
      }
    }
}

// ---------------- 5) grouped GEMM2: partial = w_row * (hidden @ Wo), fp32 out ----------------
__global__ __launch_bounds__(256) void moe_gemm2(
    const bf16_t* __restrict__ hidden, const float* __restrict__ wout,
    const float* __restrict__ wArena, const int* __restrict__ counts,
    const int* __restrict__ offsets, float* __restrict__ partial)
{
  const int e = blockIdx.z;
  const int cnt = counts[e];
  const int m0 = blockIdx.x * BM;
  if (m0 >= cnt) return;
  const int n0 = blockIdx.y * BN;
  const int base = offsets[e];
  const float* wo = wout + (size_t)e * DE * DM;

  __shared__ bf16_t As[BM * LDSA];
  __shared__ bf16_t Bs[BN * LDSB];

  const int tid = threadIdx.x;
  const int lane = tid & 31;
  const int wid = tid >> 5;
  const int wm = (wid & 3) * 32;
  const int wn = (wid >> 2) * 32;
  const int lrow = lane & 15;

  v8f c[2][2];
#pragma unroll
  for (int i = 0; i < 2; ++i)
#pragma unroll
    for (int j = 0; j < 2; ++j)
#pragma unroll
      for (int q = 0; q < 8; ++q) c[i][j][q] = 0.f;

  for (int k0 = 0; k0 < DE; k0 += BK) {
    { // stage A: bf16 arena rows, direct 16B copies
      const int row = tid >> 1;
      int m = m0 + row; if (m >= cnt) m = cnt - 1;
      const int kq = (tid & 1) * 16;
      const int4* src = reinterpret_cast<const int4*>(
          hidden + (size_t)(base + m) * DE + k0 + kq);
      int4 v0 = src[0], v1 = src[1];
      *reinterpret_cast<int4*>(&As[row * LDSA + kq]) = v0;
      *reinterpret_cast<int4*>(&As[row * LDSA + kq + 8]) = v1;
    }
#pragma unroll
    for (int i = 0; i < 2; ++i) { // stage B: fp32 [H,D] -> bf16 [N][K]
      const int u = tid + 256 * i;
      const int n = u & 63;
      const int k = (u >> 6) * 4;
      const size_t gidx = (size_t)(k0 + k) * DM + (n0 + n);
      bf16x4 h;
#pragma unroll
      for (int q = 0; q < 4; ++q) h[q] = (bf16_t)wo[gidx + (size_t)q * DM];
      *reinterpret_cast<bf16x4*>(&Bs[n * LDSB + k]) = h;
    }
    __syncthreads();

    v16bf a[2], b[2];
#pragma unroll
    for (int mi = 0; mi < 2; ++mi) a[mi] = frag_a(As, wm + mi * 16 + lrow, lane);
#pragma unroll
    for (int ni = 0; ni < 2; ++ni) b[ni] = frag_b(Bs, wn + ni * 16 + lrow, lane);
#pragma unroll
    for (int mi = 0; mi < 2; ++mi)
#pragma unroll
      for (int ni = 0; ni < 2; ++ni)
        c[mi][ni] = __builtin_amdgcn_wmma_f32_16x16x32_bf16(
            false, a[mi], false, b[ni], (short)0, c[mi][ni], false, false);
    __syncthreads();
  }

  const int mhi = (lane >> 4) * 8;
#pragma unroll
  for (int mi = 0; mi < 2; ++mi)
#pragma unroll
    for (int ni = 0; ni < 2; ++ni) {
      const int col = n0 + wn + ni * 16 + lrow;
#pragma unroll
      for (int j = 0; j < 8; ++j) {
        const int m = wm + mi * 16 + mhi + j;
        if (m0 + m < cnt) {
          const float w = wArena[base + m0 + m];
          partial[(size_t)(base + m0 + m) * DM + col] = w * c[mi][ni][j];
        }
      }
    }
}

// ---------------- 6) combine the two expert partials per token ----------------
__global__ __launch_bounds__(256) void moe_combine(
    const float* __restrict__ partial, const int* __restrict__ rowOf,
    float* __restrict__ out)
{
  const int t = blockIdx.x;
  const int r0 = rowOf[2 * t], r1 = rowOf[2 * t + 1];
  const float4* p0 = reinterpret_cast<const float4*>(partial + (size_t)r0 * DM);
  const float4* p1 = reinterpret_cast<const float4*>(partial + (size_t)r1 * DM);
  float4* o = reinterpret_cast<float4*>(out + (size_t)t * DM);
  const int i = threadIdx.x;
  float4 a = p0[i], b = p1[i];
  float4 r; r.x = a.x + b.x; r.y = a.y + b.y; r.z = a.z + b.z; r.w = a.w + b.w;
  o[i] = r;
}

extern "C" void kernel_launch(void* const* d_in, const int* in_sizes, int n_in,
                              void* d_out, int out_size, void* d_ws, size_t ws_size,
                              hipStream_t stream)
{
  const float* x  = (const float*)d_in[0];
  const float* gw = (const float*)d_in[1];
  const float* gb = (const float*)d_in[2];
  const float* wg = (const float*)d_in[3];
  const float* wv = (const float*)d_in[4];
  const float* wo = (const float*)d_in[5];
  float* out = (float*)d_out;

  char* p = (char*)d_ws;
  size_t off = 0;
  auto take = [&](size_t bytes) {
    char* r = p + off;
    off += (bytes + 255) & ~(size_t)255;
    return r;
  };
  int*    sel     = (int*)   take((size_t)2 * N_TOK * sizeof(int));
  float*  selw    = (float*) take((size_t)2 * N_TOK * sizeof(float));
  int*    counts  = (int*)   take(NEXP * sizeof(int));
  int*    offs    = (int*)   take(NEXP * sizeof(int));
  int*    tokList = (int*)   take((size_t)2 * N_TOK * sizeof(int));
  float*  wArena  = (float*) take((size_t)2 * N_TOK * sizeof(float));
  int*    rowOf   = (int*)   take((size_t)2 * N_TOK * sizeof(int));
  bf16_t* hidden  = (bf16_t*)take((size_t)2 * N_TOK * DE * sizeof(bf16_t));
  float*  partial = (float*) take((size_t)2 * N_TOK * DM * sizeof(float));

  moe_router <<<N_TOK / 8, 256, 0, stream>>>(x, gw, gb, sel, selw);
  moe_count  <<<1, 256, 0, stream>>>(sel, counts, offs);
  moe_gather <<<NEXP, 256, 0, stream>>>(sel, selw, offs, tokList, wArena, rowOf);
  moe_gemm1  <<<dim3(N_TOK / BM, DE / BN, NEXP), 256, 0, stream>>>(
      x, wg, wv, tokList, counts, offs, hidden);
  moe_gemm2  <<<dim3(N_TOK / BM, DM / BN, NEXP), 256, 0, stream>>>(
      hidden, wo, wArena, counts, offs, partial);
  moe_combine<<<N_TOK, 256, 0, stream>>>(partial, rowOf, out);
}
